// GQAAttention_66984309948668
// MI455X (gfx1250) — matrix operand order repaired and checked by reading, not compiled, not run on hardware
//
#include <hip/hip_runtime.h>
#include <hip/hip_bf16.h>
#include <math.h>

typedef __bf16 bf16;
typedef __attribute__((ext_vector_type(16))) __bf16 v16bf;
typedef __attribute__((ext_vector_type(8)))  __bf16 bf16x8;
typedef __attribute__((ext_vector_type(8)))  float  v8f;
typedef __attribute__((ext_vector_type(4)))  int    v4i;

#define HIDDEN   2048
#define NHEADS   16
#define NKV      2
#define HDIM     256
#define HALF_ROT 32
#define SEQL     4096
#define QDIM     (NHEADS * HDIM)   /* 4096 */
#define KVDIM    (NKV * HDIM)      /* 512  */
#define NEGBIG   (-1.0e9f)

#if defined(__HIP_DEVICE_COMPILE__) && __has_builtin(__builtin_amdgcn_global_load_async_to_lds_b128)
#define HAVE_ASYNC_LDS 1
#else
#define HAVE_ASYNC_LDS 0
#endif

typedef __attribute__((address_space(1))) v4i* g_v4i_t;  // global (AS1) int4*
typedef __attribute__((address_space(3))) v4i* l_v4i_t;  // LDS    (AS3) int4*

__device__ __forceinline__ g_v4i_t as_global_v4i(const void* p) {
  return (g_v4i_t)(unsigned long long)p;                  // AS1 = full 64-bit address
}
__device__ __forceinline__ l_v4i_t as_lds_v4i(const void* p) {
  return (l_v4i_t)(unsigned)(unsigned long long)p;        // AS3 = low 32 bits (LDS offset)
}

__device__ __forceinline__ void wait_async0() {
#if __has_builtin(__builtin_amdgcn_s_wait_asynccnt)
  __builtin_amdgcn_s_wait_asynccnt(0);
#elif HAVE_ASYNC_LDS
  asm volatile("s_wait_asynccnt 0" ::: "memory");
#endif
}

__device__ __forceinline__ v16bf cat8(bf16x8 lo, bf16x8 hi) {
  return __builtin_shufflevector(lo, hi, 0,1,2,3,4,5,6,7,8,9,10,11,12,13,14,15);
}

// ---------------- f32 -> bf16 converters ----------------
__global__ void cvt_f32_bf16(const float* __restrict__ src, bf16* __restrict__ dst, long n) {
  long i = (long)blockIdx.x * blockDim.x + threadIdx.x;
  long stride = (long)gridDim.x * blockDim.x;
  for (; i < n; i += stride) dst[i] = (bf16)src[i];
}

// src is (K, N) row-major f32; dst is (N, K) row-major bf16 (transposed weights)
__global__ void cvtT_f32_bf16(const float* __restrict__ src, bf16* __restrict__ dst, int K, int N) {
  long n = (long)K * N;
  long i = (long)blockIdx.x * blockDim.x + threadIdx.x;
  if (i >= n) return;
  int k  = (int)(i / N);
  int nn = (int)(i % N);
  dst[(long)nn * K + k] = (bf16)src[i];
}

// ---------------- bf16 GEMM: C(MxN) = A(MxK) @ B(KxN), B given transposed (NxK) ----
// Block: 256 threads = 8 waves. Each wave computes a 16x128 tile; block tile 128x128.
// A/B fragments load straight from global: X + all bf16 weights fit in the 192 MB L2.
__global__ __launch_bounds__(256) void gemm_bf16(
    const bf16* __restrict__ A, const bf16* __restrict__ BT,
    bf16* __restrict__ Cb, float* __restrict__ Cf,
    int M, int N, int K)
{
  const int lane = threadIdx.x & 31;
  const int wave = threadIdx.x >> 5;
  const int lr   = lane & 15;
  const int hi   = lane >> 4;
  const int m0   = blockIdx.y * 128 + wave * 16;
  const int n0   = blockIdx.x * 128;

  v8f acc[8];
#pragma unroll
  for (int t = 0; t < 8; t++)
#pragma unroll
    for (int j = 0; j < 8; j++) acc[t][j] = 0.0f;

  // A-fragment k order used here (contiguous, same as B): lanes 0-15 -> M=lane, K 0..15;
  // lanes 16-31 -> M=lane-16, K 16..31.
  const bf16* arow = A + (long)(m0 + lr) * K + (hi ? 16 : 0);

  for (int kc = 0; kc < K; kc += 32) {
    bf16x8 alo = *(const bf16x8*)(arow + kc);
    bf16x8 ahi = *(const bf16x8*)(arow + kc + 8);
    v16bf a = cat8(alo, ahi);
#pragma unroll
    for (int t = 0; t < 8; t++) {
      // B-fragment: lane<16 -> col N=n, K=0..15 ; lane>=16 -> col N=n, K=16..31
      const bf16* bp = BT + (long)(n0 + t * 16 + lr) * K + kc + (hi ? 16 : 0);
      v16bf b = *(const v16bf*)bp;
      acc[t] = __builtin_amdgcn_wmma_f32_16x16x32_bf16(
          false, a, false, b, (short)0, acc[t], false, false);
    }
  }

  // C layout: VGPR r: lanes 0-15 -> M=r, N=lane ; lanes 16-31 -> M=8+r, N=lane-16
#pragma unroll
  for (int t = 0; t < 8; t++) {
#pragma unroll
    for (int r = 0; r < 8; r++) {
      int row = m0 + r + hi * 8;
      int col = n0 + t * 16 + lr;
      if (Cf) Cf[(long)row * N + col] = acc[t][r];
      else    Cb[(long)row * N + col] = (bf16)acc[t][r];
    }
  }
}

// ---------------- partial RoPE (in place, heads 0..15 = Q, 16..17 = K) ----------
__global__ void rope_qk(bf16* __restrict__ qb, bf16* __restrict__ kb,
                        const float* __restrict__ cosT, const float* __restrict__ sinT) {
  long i = (long)blockIdx.x * blockDim.x + threadIdx.x;
  const long total = (long)SEQL * (NHEADS + NKV) * HALF_ROT;
  if (i >= total) return;
  int half = (int)(i % HALF_ROT);
  int head = (int)((i / HALF_ROT) % (NHEADS + NKV));
  int pos  = (int)(i / ((long)HALF_ROT * (NHEADS + NKV)));
  float c = cosT[pos * HALF_ROT + half];
  float s = sinT[pos * HALF_ROT + half];
  bf16* p = (head < NHEADS)
      ? qb + (long)pos * QDIM + head * HDIM
      : kb + (long)pos * KVDIM + (head - NHEADS) * HDIM;
  float x1 = (float)p[half];
  float x2 = (float)p[half + HALF_ROT];
  p[half]            = (bf16)(x1 * c - x2 * s);
  p[half + HALF_ROT] = (bf16)(x1 * s + x2 * c);
}

// ---------------- export rope'd K and V as f32 to d_out ------------------------
__global__ void export_kv(const bf16* __restrict__ kb, const bf16* __restrict__ vb,
                          float* __restrict__ outk, float* __restrict__ outv) {
  long i = (long)blockIdx.x * blockDim.x + threadIdx.x;
  const long n = (long)SEQL * KVDIM;
  if (i >= n) return;
  outk[i] = (float)kb[i];
  outv[i] = (float)vb[i];
}

// ---------------- flash attention (causal, GQA 16 q-heads over 2 kv-heads) -----
// Grid: (SEQ/64, NHEADS). Block: 128 threads = 4 waves; each wave = 16 query rows.
// K/V tiles staged via GLOBAL_LOAD_ASYNC_TO_LDS (ASYNCcnt) when available.
__global__ __launch_bounds__(128) void attn_kernel(
    const bf16* __restrict__ qb, const bf16* __restrict__ kb,
    const bf16* __restrict__ vb, bf16* __restrict__ ab)
{
  __shared__ __align__(64) bf16 k_lds[32 * HDIM];   // [key][dim], 16 KB
  __shared__ __align__(64) bf16 vT_lds[HDIM * 32];  // [dim][key], 16 KB
  __shared__ __align__(64) bf16 p_lds[4 * 16 * 32]; // per-wave [row][key], 4 KB
#if HAVE_ASYNC_LDS
  __shared__ __align__(64) bf16 v_lds[32 * HDIM];   // [key][dim] row-major, 16 KB
#endif

  const int tid  = threadIdx.x;
  const int lane = tid & 31;
  const int wave = tid >> 5;
  const int lr   = lane & 15;
  const int hi   = lane >> 4;
  const int h    = blockIdx.y;
  const int kvh  = h >> 3;              // g = 16/2 = 8
  const int q0   = blockIdx.x * 64;
  const int mrow0 = q0 + wave * 16;

  // Preload Q fragments for this wave's 16 rows: 8 chunks over HDIM=256
  v16bf qf[8];
  {
    const bf16* qrow = qb + (long)(mrow0 + lr) * QDIM + h * HDIM + (hi ? 16 : 0);
#pragma unroll
    for (int kc = 0; kc < 8; kc++) {
      bf16x8 lo = *(const bf16x8*)(qrow + kc * 32);
      bf16x8 hh = *(const bf16x8*)(qrow + kc * 32 + 8);
      qf[kc] = cat8(lo, hh);
    }
  }

  v8f o[16];
#pragma unroll
  for (int t = 0; t < 16; t++)
#pragma unroll
    for (int j = 0; j < 8; j++) o[t][j] = 0.0f;
  float m_i[8], l_i[8];
#pragma unroll
  for (int r = 0; r < 8; r++) { m_i[r] = -1.0e30f; l_i[r] = 0.0f; }

  const int nblocks = 2 * blockIdx.x + 2;   // causal: keys up to q0+63
  for (int nb = 0; nb < nblocks; nb++) {
    const int n0 = nb * 32;

    // ---- stage K and V tiles into LDS ----
#if HAVE_ASYNC_LDS
    // Async DMA path: per-lane 16-byte copies, VGPR-bypassing, tracked by ASYNCcnt.
#pragma unroll
    for (int it = 0; it < 8; it++) {
      int ch  = tid + it * 128;                  // 1024 vec8 chunks = 32 keys x 256 dims
      int key = ch >> 5;
      int d   = (ch & 31) * 8;
      const long src = (long)(n0 + key) * KVDIM + kvh * HDIM + d;
      __builtin_amdgcn_global_load_async_to_lds_b128(
          as_global_v4i(kb + src), as_lds_v4i(k_lds + key * HDIM + d), 0, 0);
      __builtin_amdgcn_global_load_async_to_lds_b128(
          as_global_v4i(vb + src), as_lds_v4i(v_lds + key * HDIM + d), 0, 0);
    }
    wait_async0();
    __syncthreads();
    // Transpose V inside LDS (overlaps with S-compute; made visible by the
    // post-softmax __syncthreads before P@V consumes vT_lds).
    for (int ch = tid; ch < 1024; ch += 128) {
      int key = ch >> 5;
      int d   = (ch & 31) * 8;
      bf16x8 vv = *(const bf16x8*)(v_lds + key * HDIM + d);
#pragma unroll
      for (int j = 0; j < 8; j++) vT_lds[(d + j) * 32 + key] = vv[j];
    }
#else
    for (int ch = tid; ch < 1024; ch += 128) {      // 1024 vec8 chunks = 32x256
      int key = ch >> 5;
      int d   = (ch & 31) * 8;
      const long src = (long)(n0 + key) * KVDIM + kvh * HDIM + d;
      bf16x8 kv = *(const bf16x8*)(kb + src);
      *(bf16x8*)(k_lds + key * HDIM + d) = kv;
      bf16x8 vv = *(const bf16x8*)(vb + src);
#pragma unroll
      for (int j = 0; j < 8; j++) vT_lds[(d + j) * 32 + key] = vv[j];
    }
    __syncthreads();
#endif

    // ---- S = Q @ K^T  (16 x 32) ----
    v8f s[2];
#pragma unroll
    for (int nt = 0; nt < 2; nt++)
#pragma unroll
      for (int j = 0; j < 8; j++) s[nt][j] = 0.0f;
#pragma unroll
    for (int nt = 0; nt < 2; nt++) {
#pragma unroll
      for (int kc = 0; kc < 8; kc++) {
        const v16bf b = *(const v16bf*)(k_lds + (nt * 16 + lr) * HDIM + kc * 32 + hi * 16);
        s[nt] = __builtin_amdgcn_wmma_f32_16x16x32_bf16(
            false, qf[kc], false, b, (short)0, s[nt], false, false);
      }
    }

    // ---- scale + causal mask + online softmax ----
#pragma unroll
    for (int r = 0; r < 8; r++) {
      const int row = mrow0 + r + hi * 8;
      float v0 = s[0][r] * 0.0625f + ((n0 + lr)      > row ? NEGBIG : 0.0f);
      float v1 = s[1][r] * 0.0625f + ((n0 + 16 + lr) > row ? NEGBIG : 0.0f);
      float rm = fmaxf(v0, v1);
#pragma unroll
      for (int off = 1; off < 16; off <<= 1) rm = fmaxf(rm, __shfl_xor(rm, off, 16));
      float mnew = fmaxf(m_i[r], rm);
      float corr = __expf(m_i[r] - mnew);
      m_i[r] = mnew;
      float p0 = __expf(v0 - mnew);
      float p1 = __expf(v1 - mnew);
      float rs = p0 + p1;
#pragma unroll
      for (int off = 1; off < 16; off <<= 1) rs += __shfl_xor(rs, off, 16);
      l_i[r] = l_i[r] * corr + rs;
#pragma unroll
      for (int dt = 0; dt < 16; dt++) o[dt][r] *= corr;
      // bounce P through LDS to convert C-layout -> A-layout
      p_lds[wave * 512 + (r + hi * 8) * 32 + lr]      = (bf16)p0;
      p_lds[wave * 512 + (r + hi * 8) * 32 + 16 + lr] = (bf16)p1;
    }
    __syncthreads();

    // ---- O += P @ V ----
    {
      const bf16* pp = p_lds + wave * 512 + lr * 32 + (hi ? 16 : 0);
      bf16x8 plo = *(const bf16x8*)pp;
      bf16x8 phi = *(const bf16x8*)(pp + 8);
      v16bf pa = cat8(plo, phi);
#pragma unroll
      for (int dt = 0; dt < 16; dt++) {
        const v16bf b = *(const v16bf*)(vT_lds + (dt * 16 + lr) * 32 + hi * 16);
        o[dt] = __builtin_amdgcn_wmma_f32_16x16x32_bf16(
            false, pa, false, b, (short)0, o[dt], false, false);
      }
    }
    __syncthreads();
  }

  // ---- epilogue: normalize and write context (bf16) for the Wo GEMM ----
#pragma unroll
  for (int dt = 0; dt < 16; dt++) {
#pragma unroll
    for (int r = 0; r < 8; r++) {
      const int row = mrow0 + r + hi * 8;
      float val = o[dt][r] / l_i[r];
      ab[(long)row * QDIM + h * HDIM + dt * 16 + lr] = (bf16)val;
    }
  }
}

// ---------------------------------------------------------------------------
extern "C" void kernel_launch(void* const* d_in, const int* in_sizes, int n_in,
                              void* d_out, int out_size, void* d_ws, size_t ws_size,
                              hipStream_t stream) {
  const float* x    = (const float*)d_in[0];
  const float* cosT = (const float*)d_in[1];
  const float* sinT = (const float*)d_in[2];
  // d_in[3] = attention_mask: exactly causal triu * -1e9; handled analytically.
  const float* wq   = (const float*)d_in[4];
  const float* wk   = (const float*)d_in[5];
  const float* wv   = (const float*)d_in[6];
  const float* wo   = (const float*)d_in[7];

  float* out  = (float*)d_out;                 // (4096, 2048)
  float* outk = out  + (long)SEQL * HIDDEN;    // (4096, 2, 256)
  float* outv = outk + (long)SEQL * KVDIM;     // (4096, 2, 256)

  char* ws = (char*)d_ws;
  size_t off = 0;
  auto alloc = [&](size_t bytes) -> void* {
    void* p = ws + off;
    off += (bytes + 255) & ~(size_t)255;
    return p;
  };
  bf16* xb  = (bf16*)alloc((size_t)SEQL * HIDDEN * 2);
  bf16* wqT = (bf16*)alloc((size_t)HIDDEN * QDIM * 2);
  bf16* wkT = (bf16*)alloc((size_t)HIDDEN * KVDIM * 2);
  bf16* wvT = (bf16*)alloc((size_t)HIDDEN * KVDIM * 2);
  bf16* woT = (bf16*)alloc((size_t)QDIM * HIDDEN * 2);
  bf16* qb  = (bf16*)alloc((size_t)SEQL * QDIM * 2);
  bf16* kb  = (bf16*)alloc((size_t)SEQL * KVDIM * 2);
  bf16* vb  = (bf16*)alloc((size_t)SEQL * KVDIM * 2);
  bf16* ab  = (bf16*)alloc((size_t)SEQL * QDIM * 2);

  // 1) precision conversion (weights transposed for WMMA-B contiguity)
  cvt_f32_bf16<<<2048, 256, 0, stream>>>(x, xb, (long)SEQL * HIDDEN);
  cvtT_f32_bf16<<<(HIDDEN * QDIM + 255) / 256, 256, 0, stream>>>(wq, wqT, HIDDEN, QDIM);
  cvtT_f32_bf16<<<(HIDDEN * KVDIM + 255) / 256, 256, 0, stream>>>(wk, wkT, HIDDEN, KVDIM);
  cvtT_f32_bf16<<<(HIDDEN * KVDIM + 255) / 256, 256, 0, stream>>>(wv, wvT, HIDDEN, KVDIM);
  cvtT_f32_bf16<<<(QDIM * HIDDEN + 255) / 256, 256, 0, stream>>>(wo, woT, QDIM, HIDDEN);

  // 2) QKV projections (WMMA bf16, f32 accumulate)
  gemm_bf16<<<dim3(QDIM / 128,  SEQL / 128), 256, 0, stream>>>(xb, wqT, qb, nullptr, SEQL, QDIM,  HIDDEN);
  gemm_bf16<<<dim3(KVDIM / 128, SEQL / 128), 256, 0, stream>>>(xb, wkT, kb, nullptr, SEQL, KVDIM, HIDDEN);
  gemm_bf16<<<dim3(KVDIM / 128, SEQL / 128), 256, 0, stream>>>(xb, wvT, vb, nullptr, SEQL, KVDIM, HIDDEN);

  // 3) partial RoPE on Q and K, then export new_k / new_v (f32)
  {
    long total = (long)SEQL * (NHEADS + NKV) * HALF_ROT;
    rope_qk<<<(unsigned)((total + 255) / 256), 256, 0, stream>>>(qb, kb, cosT, sinT);
  }
  export_kv<<<(SEQL * KVDIM + 255) / 256, 256, 0, stream>>>(kb, vb, outk, outv);

  // 4) causal flash attention
  attn_kernel<<<dim3(SEQL / 64, NHEADS), 128, 0, stream>>>(qb, kb, vb, ab);

  // 5) output projection -> f32 d_out
  gemm_bf16<<<dim3(HIDDEN / 128, SEQL / 128), 256, 0, stream>>>(ab, woT, nullptr, out, SEQL, HIDDEN, QDIM);
}
